// STiSAN_4999341932717
// MI455X (gfx1250) — matrix operand order repaired and checked by reading, not compiled, not run on hardware
//
#include <hip/hip_runtime.h>
#include <hip/hip_bf16.h>
#include <stdint.h>

typedef __attribute__((ext_vector_type(16))) _Float16 v16h;
typedef __attribute__((ext_vector_type(8)))  float    v8f;

#define B_   512
#define L_   100
#define LT_  101
#define LP_  128
#define D_   256
#define DH_  512
#define LG_  12

__device__ __forceinline__ unsigned short f2h(float f){
  union { _Float16 h; unsigned short u; } c; c.h = (_Float16)f; return c.u;
}
__device__ __forceinline__ float wredsum(float v){
#pragma unroll
  for (int o=16;o>0;o>>=1) v += __shfl_xor(v,o,32);
  return v;
}
__device__ __forceinline__ float wredmax(float v){
#pragma unroll
  for (int o=16;o>0;o>>=1) v = fmaxf(v, __shfl_xor(v,o,32));
  return v;
}

// ---------------------------------------------------------------------------
// Generic NT GEMM:  C[M x N] = A[M x K] * B[N x K]^T  (+bias, optional relu)
// One wave computes a 32x32 output tile: 2 A-frags x 2 B-frags -> 4 independent
// v_wmma_f32_16x16x32_f16 per K-step (no D->A/B RAW between them, so they
// pipeline; per-WMMA global load traffic is halved vs 16x16 tiling).
// A, B are f16 (as u16), row-major with leading dims lda/ldb (even).
// ---------------------------------------------------------------------------
__global__ void gemm_nt(const unsigned short* __restrict__ A, long long sA, int lda,
                        const unsigned short* __restrict__ Bm, long long sB, int ldb,
                        float* __restrict__ Cf, long long sCf, int ldc,
                        unsigned short* __restrict__ Cb, long long sCb, int ldcb,
                        const float* __restrict__ bias,
                        int M, int N, int K, int relu)
{
  const int bb   = blockIdx.z;
  const int lane = threadIdx.x;
  const int row0 = blockIdx.y * 32;
  const int col0 = blockIdx.x * 32;
  const unsigned int* Ap = (const unsigned int*)(A + (size_t)bb * sA);
  const unsigned int* Bp = (const unsigned int*)(Bm + (size_t)bb * sB);
  int am[2], bn[2];
  am[0] = row0      + (lane & 15); if (am[0] >= M) am[0] = M-1;  // clamped rows never stored
  am[1] = row0 + 16 + (lane & 15); if (am[1] >= M) am[1] = M-1;
  bn[0] = col0      + (lane & 15); if (bn[0] >= N) bn[0] = N-1;
  bn[1] = col0 + 16 + (lane & 15); if (bn[1] >= N) bn[1] = N-1;
  const int akb = (lane >> 4) * 8;    // A: hi half-wave holds K+8 / K+24 blocks
  const int bkb = (lane >> 4) * 8;    // B: hi half-wave holds K+8 / K+24 blocks
  v8f acc[2][2];
#pragma unroll
  for (int ai=0; ai<2; ai++)
#pragma unroll
    for (int bi=0; bi<2; bi++)
      acc[ai][bi] = (v8f){0.f,0.f,0.f,0.f,0.f,0.f,0.f,0.f};

  for (int k0 = 0; k0 < K; k0 += 32){
    union { unsigned int u[8]; v16h v; } af[2], bfm[2];
#pragma unroll
    for (int t=0; t<2; t++){
      const int abase = (am[t] * lda + k0) >> 1;   // u32 units
      const int bbase = (bn[t] * ldb + k0) >> 1;
#pragma unroll
      for (int i=0;i<8;i++){
        const int koff = (i < 4) ? (akb + 2*i) : (16 + akb + 2*(i-4));
        af[t].u[i]  = Ap[abase + (koff >> 1)];
        bfm[t].u[i] = Bp[bbase + (((i < 4) ? (bkb + 2*i) : (16 + bkb + 2*(i-4))) >> 1)];
      }
    }
    acc[0][0] = __builtin_amdgcn_wmma_f32_16x16x32_f16(false, af[0].v, false, bfm[0].v,
                                                       (short)0, acc[0][0], false, false);
    acc[0][1] = __builtin_amdgcn_wmma_f32_16x16x32_f16(false, af[0].v, false, bfm[1].v,
                                                       (short)0, acc[0][1], false, false);
    acc[1][0] = __builtin_amdgcn_wmma_f32_16x16x32_f16(false, af[1].v, false, bfm[0].v,
                                                       (short)0, acc[1][0], false, false);
    acc[1][1] = __builtin_amdgcn_wmma_f32_16x16x32_f16(false, af[1].v, false, bfm[1].v,
                                                       (short)0, acc[1][1], false, false);
  }

  const int rbase = ((lane >> 4) << 3);
#pragma unroll
  for (int bi=0; bi<2; bi++){
    const int colc = col0 + bi*16 + (lane & 15);
    float bv = 0.f;
    if (bias && colc < N) bv = bias[colc];
#pragma unroll
    for (int ai=0; ai<2; ai++){
      const int rb = row0 + ai*16 + rbase;
#pragma unroll
      for (int r=0;r<8;r++){
        const int row = rb + r;
        if (row < M && colc < N){
          float v = acc[ai][bi][r] + bv;
          if (relu) v = fmaxf(v, 0.f);
          if (Cf) Cf[(size_t)bb*sCf + (size_t)row*ldc  + colc] = v;
          if (Cb) Cb[(size_t)bb*sCb + (size_t)row*ldcb + colc] = f2h(v);
        }
      }
    }
  }
}

// ---------------------------------------------------------------------------
// Weight convert + transpose: src f32 [K x N] -> dst f16 [N x K]
// ---------------------------------------------------------------------------
__global__ void convert_w(const float* __restrict__ src, unsigned short* __restrict__ dst,
                          int K, int N)
{
  int t = blockIdx.x * 256 + threadIdx.x;
  if (t < K*N){ int k = t / N, n = t - k*N; dst[(size_t)n*K + k] = f2h(src[t]); }
}

// ---------------------------------------------------------------------------
// Geo encoder: one block (64 threads) per length-12 / dim-64 sequence.
// 2 layers of projless self-attention + FFN(64->128->64), mean over rows.
// Writes the 64-wide result into out[seq*row_stride + col_off + d].
// ---------------------------------------------------------------------------
__global__ void geo_kernel(const int* __restrict__ idx, const float* __restrict__ emb,
                           const float* __restrict__ w1, const float* __restrict__ b1,
                           const float* __restrict__ w2, const float* __restrict__ b2,
                           const float* __restrict__ g1, const float* __restrict__ bb1,
                           const float* __restrict__ g2, const float* __restrict__ bb2,
                           float* __restrict__ out, int row_stride, int col_off)
{
  __shared__ float x[LG_][64];
  __shared__ float h[LG_][128];
  __shared__ float c[LG_][64];
  __shared__ float s[LG_][LG_];
  __shared__ float mu[LG_], iv[LG_];
  const int seq = blockIdx.x;
  const int t = threadIdx.x;                 // 0..63
  const int* id = idx + (size_t)seq * LG_;
  for (int r=0;r<LG_;r++){
    float pe;
    if (t < 32) pe = __sinf((float)r * __expf((float)t      * -0.28782313662425572f));
    else        pe = __cosf((float)r * __expf((float)(t-32) * -0.28782313662425572f));
    x[r][t] = emb[(size_t)id[r]*64 + t] + pe;
  }
  __syncthreads();
  for (int li=0; li<2; li++){
    const float* W1  = w1 + (size_t)li*64*128; const float* Bb1 = b1 + (size_t)li*128;
    const float* W2  = w2 + (size_t)li*128*64; const float* Bb2 = b2 + (size_t)li*64;
    const float* G1  = g1 + (size_t)li*64;     const float* B1g = bb1 + (size_t)li*64;
    const float* G2  = g2 + (size_t)li*64;     const float* B2g = bb2 + (size_t)li*64;
    for (int e=t; e<LG_*LG_; e+=64){
      int q = e / LG_, k = e - q*LG_;
      float a = 0.f;
      for (int d=0; d<64; d++) a += x[q][d]*x[k][d];
      s[q][k] = a * 0.125f;
    }
    __syncthreads();
    if (t < LG_){
      float m = -1e30f; for (int k=0;k<LG_;k++) m = fmaxf(m, s[t][k]);
      float sm = 0.f;   for (int k=0;k<LG_;k++){ float p = __expf(s[t][k]-m); s[t][k]=p; sm+=p; }
      float inv = 1.f/sm; for (int k=0;k<LG_;k++) s[t][k] *= inv;
    }
    __syncthreads();
    for (int r=0;r<LG_;r++){
      float a = 0.f;
      for (int k=0;k<LG_;k++) a += s[r][k]*x[k][t];
      c[r][t] = a + x[r][t];
    }
    __syncthreads();
    if (t < LG_){
      float m = 0.f; for (int d=0;d<64;d++) m += c[t][d]; m *= (1.f/64.f);
      float v = 0.f; for (int d=0;d<64;d++){ float u=c[t][d]-m; v+=u*u; } v *= (1.f/64.f);
      mu[t]=m; iv[t]=rsqrtf(v+1e-5f);
    }
    __syncthreads();
    for (int r=0;r<LG_;r++) x[r][t] = (c[r][t]-mu[r])*iv[r]*G1[t] + B1g[t];
    __syncthreads();
    for (int r=0;r<LG_;r++){
      for (int j=t; j<128; j+=64){
        float a = Bb1[j];
        for (int d=0; d<64; d++) a += x[r][d]*W1[d*128+j];
        h[r][j] = fmaxf(a, 0.f);
      }
    }
    __syncthreads();
    for (int r=0;r<LG_;r++){
      float a = Bb2[t];
      for (int j=0;j<128;j++) a += h[r][j]*W2[j*64+t];
      c[r][t] = a + x[r][t];
    }
    __syncthreads();
    if (t < LG_){
      float m = 0.f; for (int d=0;d<64;d++) m += c[t][d]; m *= (1.f/64.f);
      float v = 0.f; for (int d=0;d<64;d++){ float u=c[t][d]-m; v+=u*u; } v *= (1.f/64.f);
      mu[t]=m; iv[t]=rsqrtf(v+1e-5f);
    }
    __syncthreads();
    for (int r=0;r<LG_;r++) x[r][t] = (c[r][t]-mu[r])*iv[r]*G2[t] + B2g[t];
    __syncthreads();
  }
  float a = 0.f;
  for (int r=0;r<LG_;r++) a += x[r][t];
  out[(size_t)seq*row_stride + col_off + t] = a * (1.f/(float)LG_);
}

// src = concat(emb_poi, geo) + sin_pe(pos, 256); also f16 copy
__global__ void finalize_src(const int* __restrict__ poi, const float* __restrict__ embp,
                             const float* __restrict__ stime,
                             float* __restrict__ X, unsigned short* __restrict__ Xh)
{
  const int row = blockIdx.x;              // b*L + l
  const int b = row / L_;
  const int d = threadIdx.x;
  const float t0 = stime[b*L_], tN = stime[b*L_ + L_-1], tt = stime[row];
  const float pos = (tt - t0) / (tN - t0 + 1e-9f) * (float)(L_-1);
  float pe;
  if (d < 128) pe = __sinf(pos * __expf((float)d       * -0.07195578415606392f));
  else         pe = __cosf(pos * __expf((float)(d-128) * -0.07195578415606392f));
  float base;
  if (d < 192) base = embp[(size_t)poi[row]*192 + d];
  else         base = X[(size_t)row*D_ + d];           // geo part already there
  const float v = base + pe;
  X [(size_t)row*D_ + d] = v;
  Xh[(size_t)row*D_ + d] = f2h(v);
}

__global__ void finalize_tgt(const int* __restrict__ poi, const float* __restrict__ embp,
                             float* __restrict__ X)
{
  const int row = blockIdx.x;
  const int d = threadIdx.x;
  if (d < 192) X[(size_t)row*D_ + d] = embp[(size_t)poi[row]*192 + d];
}

// per-batch max of min(tm,0.5) and min(gm,0.5)
__global__ void relmax(const float* __restrict__ tm, const float* __restrict__ gm,
                       float* __restrict__ tmax, float* __restrict__ gmax)
{
  __shared__ float st[256], sg[256];
  const int b = blockIdx.x, t = threadIdx.x;
  const float* tp = tm + (size_t)b * L_ * L_;
  const float* gp = gm + (size_t)b * L_ * L_;
  float mt = -1e30f, mg = -1e30f;
  for (int i = t; i < L_*L_; i += 256){
    mt = fmaxf(mt, fminf(tp[i], 0.5f));
    mg = fmaxf(mg, fminf(gp[i], 0.5f));
  }
  st[t]=mt; sg[t]=mg; __syncthreads();
  for (int o=128;o>0;o>>=1){
    if (t<o){ st[t]=fmaxf(st[t],st[t+o]); sg[t]=fmaxf(sg[t],sg[t+o]); }
    __syncthreads();
  }
  if (t==0){ tmax[b]=st[0]; gmax[b]=sg[0]; }
}

// softmax over keys with rel-bias + causal mask; writes f16 probs (zero-padded to LP)
__global__ void enc_softmax(const float* __restrict__ S, const float* __restrict__ tm,
                            const float* __restrict__ gm, const float* __restrict__ tmax,
                            const float* __restrict__ gmax, unsigned short* __restrict__ Ah)
{
  const int row = blockIdx.x;              // b*L + q
  const int b = row / L_;
  const int q = row - b*L_;
  const int lane = threadIdx.x;
  const float tmx = tmax[b], gmx = gmax[b];
  const float* srow = S  + (size_t)row * LP_;
  const float* trow = tm + (size_t)b*L_*L_ + (size_t)q*L_;
  const float* grow = gm + (size_t)b*L_*L_ + (size_t)q*L_;
  float val[4];
  float m = -1e30f;
#pragma unroll
  for (int j=0;j<4;j++){
    const int k = lane + j*32;
    float v = -1e30f;
    if (k < L_ && k <= q)
      v = srow[k]*0.0625f + (tmx - fminf(trow[k],0.5f)) + (gmx - fminf(grow[k],0.5f));
    val[j] = v; m = fmaxf(m, v);
  }
  m = wredmax(m);
  float s = 0.f;
#pragma unroll
  for (int j=0;j<4;j++){
    const int k = lane + j*32;
    float p = 0.f;
    if (k < L_ && k <= q) p = __expf(val[j] - m);
    val[j] = p; s += p;
  }
  s = wredsum(s);
  const float inv = 1.f / s;
  unsigned short* arow = Ah + (size_t)row * LP_;
#pragma unroll
  for (int j=0;j<4;j++) arow[lane + j*32] = f2h(val[j] * inv);
}

// x = LN(x + add); also f16 copy. One wave per row.
__global__ void add_ln(float* __restrict__ X, const float* __restrict__ Add,
                       const float* __restrict__ g, const float* __restrict__ bta,
                       unsigned short* __restrict__ Xh)
{
  const int row = blockIdx.x;
  const int lane = threadIdx.x;
  float* xr = X + (size_t)row * D_;
  const float* ar = Add + (size_t)row * D_;
  float v[8]; float s = 0.f;
#pragma unroll
  for (int i=0;i<8;i++){ const int d = lane + i*32; v[i] = xr[d] + ar[d]; s += v[i]; }
  s = wredsum(s); const float mu = s * (1.f/D_);
  float var = 0.f;
#pragma unroll
  for (int i=0;i<8;i++){ const float u = v[i]-mu; var += u*u; }
  var = wredsum(var) * (1.f/D_);
  const float inv = rsqrtf(var + 1e-5f);
#pragma unroll
  for (int i=0;i<8;i++){
    const int d = lane + i*32;
    const float o = (v[i]-mu)*inv*g[d] + bta[d];
    xr[d] = o; Xh[(size_t)row*D_ + d] = f2h(o);
  }
}

__global__ void extract_mem(const float* __restrict__ src, float* __restrict__ mem,
                            unsigned short* __restrict__ memh)
{
  const int b = blockIdx.x, d = threadIdx.x;
  const float v = src[((size_t)b*L_ + (L_-1))*D_ + d];
  mem [(size_t)b*D_ + d] = v;
  memh[(size_t)b*D_ + d] = f2h(v);
}

// x=LN(tgt+cdec,dec_ln); out=LN(x+mem,norm); res=dot(out,tgt). One wave per row.
__global__ void decoder_out(const float* __restrict__ tgt, const float* __restrict__ cdec,
                            const float* __restrict__ mem,
                            const float* __restrict__ dg, const float* __restrict__ db,
                            const float* __restrict__ ng, const float* __restrict__ nb,
                            float* __restrict__ out)
{
  const int row = blockIdx.x;             // b*LT + t
  const int b = row / LT_;
  const int lane = threadIdx.x;
  float tv[8], xv[8];
  float s = 0.f;
#pragma unroll
  for (int i=0;i<8;i++){
    const int d = lane + i*32;
    tv[i] = tgt[(size_t)row*D_ + d];
    xv[i] = tv[i] + cdec[(size_t)b*D_ + d];
    s += xv[i];
  }
  s = wredsum(s); const float mu = s*(1.f/D_);
  float var = 0.f;
#pragma unroll
  for (int i=0;i<8;i++){ const float u = xv[i]-mu; var += u*u; }
  var = wredsum(var)*(1.f/D_);
  const float inv = rsqrtf(var+1e-5f);
  float s2 = 0.f;
#pragma unroll
  for (int i=0;i<8;i++){
    const int d = lane + i*32;
    xv[i] = (xv[i]-mu)*inv*dg[d] + db[d] + mem[(size_t)b*D_ + d];
    s2 += xv[i];
  }
  s2 = wredsum(s2); const float mu2 = s2*(1.f/D_);
  float var2 = 0.f;
#pragma unroll
  for (int i=0;i<8;i++){ const float u = xv[i]-mu2; var2 += u*u; }
  var2 = wredsum(var2)*(1.f/D_);
  const float inv2 = rsqrtf(var2+1e-5f);
  float acc = 0.f;
#pragma unroll
  for (int i=0;i<8;i++){
    const int d = lane + i*32;
    acc += ((xv[i]-mu2)*inv2*ng[d] + nb[d]) * tv[i];
  }
  acc = wredsum(acc);
  if (lane == 0) out[row] = acc;
}

// ---------------------------------------------------------------------------
extern "C" void kernel_launch(void* const* d_in, const int* in_sizes, int n_in,
                              void* d_out, int out_size, void* d_ws, size_t ws_size,
                              hipStream_t stream)
{
  const int*   src_poi     = (const int*)  d_in[0];
  const int*   src_quadkey = (const int*)  d_in[1];
  const float* src_time    = (const float*)d_in[2];
  const float* tm          = (const float*)d_in[3];
  const float* gm          = (const float*)d_in[4];
  const int*   tgt_poi     = (const int*)  d_in[7];
  const int*   tgt_quadkey = (const int*)  d_in[8];
  const float* emb_poi_w   = (const float*)d_in[12];
  const float* emb_qk_w    = (const float*)d_in[13];
  const float* geo_w1      = (const float*)d_in[14];
  const float* geo_b1      = (const float*)d_in[15];
  const float* geo_w2      = (const float*)d_in[16];
  const float* geo_b2      = (const float*)d_in[17];
  const float* geo_ln1g    = (const float*)d_in[18];
  const float* geo_ln1b    = (const float*)d_in[19];
  const float* geo_ln2g    = (const float*)d_in[20];
  const float* geo_ln2b    = (const float*)d_in[21];
  const float* enc_wq      = (const float*)d_in[22];
  const float* enc_wk      = (const float*)d_in[23];
  const float* enc_wv      = (const float*)d_in[24];
  const float* enc_w1      = (const float*)d_in[25];
  const float* enc_b1      = (const float*)d_in[26];
  const float* enc_w2      = (const float*)d_in[27];
  const float* enc_b2      = (const float*)d_in[28];
  const float* enc_ln1g    = (const float*)d_in[29];
  const float* enc_ln1b    = (const float*)d_in[30];
  const float* enc_ln2g    = (const float*)d_in[31];
  const float* enc_ln2b    = (const float*)d_in[32];
  const float* dec_wv      = (const float*)d_in[35];
  const float* dec_lng     = (const float*)d_in[36];
  const float* dec_lnb     = (const float*)d_in[37];
  const float* norm_g      = (const float*)d_in[38];
  const float* norm_b      = (const float*)d_in[39];
  float* out = (float*)d_out;
  (void)in_sizes; (void)n_in; (void)out_size; (void)ws_size;

  size_t off = 0;
  auto alloc = [&](size_t bytes)->char*{
    char* p = (char*)d_ws + off;
    off += (bytes + 255) & ~(size_t)255;
    return p;
  };
  float*          src_f = (float*)         alloc((size_t)B_*L_*D_*4);
  float*          tgt_f = (float*)         alloc((size_t)B_*LT_*D_*4);
  unsigned short* src_h = (unsigned short*)alloc((size_t)B_*L_*D_*2);
  unsigned short* q_h   = (unsigned short*)alloc((size_t)B_*L_*D_*2);
  unsigned short* k_h   = (unsigned short*)alloc((size_t)B_*L_*D_*2);
  unsigned short* vt_h  = (unsigned short*)alloc((size_t)B_*D_*LP_*2);
  float*          s_f   = (float*)         alloc((size_t)B_*L_*LP_*4);
  unsigned short* a_h   = (unsigned short*)alloc((size_t)B_*L_*LP_*2);
  float*          c_f   = (float*)         alloc((size_t)B_*L_*D_*4);
  unsigned short* h_h   = (unsigned short*)alloc((size_t)B_*L_*DH_*2);
  unsigned short* wq_t  = (unsigned short*)alloc((size_t)2*D_*D_*2);
  unsigned short* wk_t  = (unsigned short*)alloc((size_t)2*D_*D_*2);
  unsigned short* wv_t  = (unsigned short*)alloc((size_t)2*D_*D_*2);
  unsigned short* w1_t  = (unsigned short*)alloc((size_t)2*D_*DH_*2);
  unsigned short* w2_t  = (unsigned short*)alloc((size_t)2*DH_*D_*2);
  unsigned short* dwv_t = (unsigned short*)alloc((size_t)D_*D_*2);
  float* tmaxb = (float*)alloc((size_t)B_*4);
  float* gmaxb = (float*)alloc((size_t)B_*4);
  float* mem_f = (float*)alloc((size_t)B_*D_*4);
  unsigned short* mem_h = (unsigned short*)alloc((size_t)B_*D_*2);
  float* cdec  = (float*)alloc((size_t)B_*D_*4);

  // zero Vt pads (cols 100..127 act as K-padding for the A@V GEMM)
  hipMemsetAsync(vt_h, 0, (size_t)B_*D_*LP_*2, stream);

  auto conv = [&](const float* s, unsigned short* dte, int K, int N){
    const int total = K*N;
    convert_w<<<(total+255)/256, 256, 0, stream>>>(s, dte, K, N);
  };
  for (int i=0;i<2;i++){
    conv(enc_wq + (size_t)i*D_*D_,  wq_t + (size_t)i*D_*D_,  D_,  D_);
    conv(enc_wk + (size_t)i*D_*D_,  wk_t + (size_t)i*D_*D_,  D_,  D_);
    conv(enc_wv + (size_t)i*D_*D_,  wv_t + (size_t)i*D_*D_,  D_,  D_);
    conv(enc_w1 + (size_t)i*D_*DH_, w1_t + (size_t)i*D_*DH_, D_,  DH_);
    conv(enc_w2 + (size_t)i*DH_*D_, w2_t + (size_t)i*DH_*D_, DH_, D_);
  }
  conv(dec_wv, dwv_t, D_, D_);

  // geo encodings written straight into column 192.. of src/tgt
  geo_kernel<<<B_*L_, 64, 0, stream>>>(src_quadkey, emb_qk_w, geo_w1, geo_b1, geo_w2, geo_b2,
                                       geo_ln1g, geo_ln1b, geo_ln2g, geo_ln2b, src_f, D_, 192);
  geo_kernel<<<B_*LT_, 64, 0, stream>>>(tgt_quadkey, emb_qk_w, geo_w1, geo_b1, geo_w2, geo_b2,
                                        geo_ln1g, geo_ln1b, geo_ln2g, geo_ln2b, tgt_f, D_, 192);
  finalize_src<<<B_*L_, 256, 0, stream>>>(src_poi, emb_poi_w, src_time, src_f, src_h);
  finalize_tgt<<<B_*LT_, 256, 0, stream>>>(tgt_poi, emb_poi_w, tgt_f);
  relmax<<<B_, 256, 0, stream>>>(tm, gm, tmaxb, gmaxb);

  const dim3 blk(32,1,1);
  for (int i=0;i<2;i++){
    // Q = src @ Wq  (f16 out)   M=100 -> 4 row tiles of 32, N=256 -> 8 col tiles
    gemm_nt<<<dim3(8,4,B_), blk, 0, stream>>>(src_h, (long long)L_*D_, D_,
        wq_t + (size_t)i*D_*D_, 0, D_,
        (float*)nullptr, 0, 0, q_h, (long long)L_*D_, D_,
        nullptr, L_, D_, D_, 0);
    // K = src @ Wk
    gemm_nt<<<dim3(8,4,B_), blk, 0, stream>>>(src_h, (long long)L_*D_, D_,
        wk_t + (size_t)i*D_*D_, 0, D_,
        (float*)nullptr, 0, 0, k_h, (long long)L_*D_, D_,
        nullptr, L_, D_, D_, 0);
    // V^T = Wv^T @ src^T  -> [D x LP], zero-padded cols.  M=256, N=100
    gemm_nt<<<dim3(4,8,B_), blk, 0, stream>>>(wv_t + (size_t)i*D_*D_, 0, D_,
        src_h, (long long)L_*D_, D_,
        (float*)nullptr, 0, 0, vt_h, (long long)D_*LP_, LP_,
        nullptr, D_, L_, D_, 0);
    // S = Q K^T  (f32, unscaled; scale folded into softmax)  M=N=100
    gemm_nt<<<dim3(4,4,B_), blk, 0, stream>>>(q_h, (long long)L_*D_, D_,
        k_h, (long long)L_*D_, D_,
        s_f, (long long)L_*LP_, LP_, (unsigned short*)nullptr, 0, 0,
        nullptr, L_, L_, D_, 0);
    enc_softmax<<<B_*L_, 32, 0, stream>>>(s_f, tm, gm, tmaxb, gmaxb, a_h);
    // C = A V  via A[L x LP] * Vt[D x LP]^T   M=100, N=256, K=128
    gemm_nt<<<dim3(8,4,B_), blk, 0, stream>>>(a_h, (long long)L_*LP_, LP_,
        vt_h, (long long)D_*LP_, LP_,
        c_f, (long long)L_*D_, D_, (unsigned short*)nullptr, 0, 0,
        nullptr, L_, D_, LP_, 0);
    add_ln<<<B_*L_, 32, 0, stream>>>(src_f, c_f,
        enc_ln1g + (size_t)i*D_, enc_ln1b + (size_t)i*D_, src_h);
    // H = relu(src W1 + b1)  (f16 out)  M=100, N=512
    gemm_nt<<<dim3(16,4,B_), blk, 0, stream>>>(src_h, (long long)L_*D_, D_,
        w1_t + (size_t)i*D_*DH_, 0, D_,
        (float*)nullptr, 0, 0, h_h, (long long)L_*DH_, DH_,
        enc_b1 + (size_t)i*DH_, L_, DH_, D_, 1);
    // F = H W2 + b2  (f32 into c_f)  M=100, N=256, K=512
    gemm_nt<<<dim3(8,4,B_), blk, 0, stream>>>(h_h, (long long)L_*DH_, DH_,
        w2_t + (size_t)i*DH_*D_, 0, DH_,
        c_f, (long long)L_*D_, D_, (unsigned short*)nullptr, 0, 0,
        enc_b2 + (size_t)i*D_, L_, D_, DH_, 0);
    add_ln<<<B_*L_, 32, 0, stream>>>(src_f, c_f,
        enc_ln2g + (size_t)i*D_, enc_ln2b + (size_t)i*D_, src_h);
  }

  extract_mem<<<B_, 256, 0, stream>>>(src_f, mem_f, mem_h);
  // decoder attention collapses exactly to cdec = mem @ dec_wv (keys identical => uniform softmax)
  // M = B = 512 -> 16 row tiles, N = 256 -> 8 col tiles
  gemm_nt<<<dim3(8,16,1), blk, 0, stream>>>(mem_h, 0, D_,
      dwv_t, 0, D_,
      cdec, 0, D_, (unsigned short*)nullptr, 0, 0,
      nullptr, B_, D_, D_, 0);
  decoder_out<<<B_*LT_, 32, 0, stream>>>(tgt_f, cdec, mem_f, dec_lng, dec_lnb,
                                         norm_g, norm_b, out);
}
